// GCN_12807592477476
// MI455X (gfx1250) — matrix operand order repaired and checked by reading, not compiled, not run on hardware
//
#include <hip/hip_runtime.h>

// ---------------------------------------------------------------------------
// GCN forward for MI455X (gfx1250, wave32, WMMA).
// f32 -> bf16 once, all GEMMs via v_wmma_f32_16x16x32_bf16 (f32 accumulate).
// Workload is ~570 FLOP/byte -> compute bound; kernel is built to keep the
// matrix pipe busy: 64x64 per-wave tile => 16 WMMA per 32-K step against
// 16 ds_load_b128, double-buffered LDS => 1 barrier per K-step, next-tile
// global loads issued before the WMMA burst.
// ---------------------------------------------------------------------------

typedef __bf16 bf16;
typedef __attribute__((ext_vector_type(4)))  __bf16 v4bf;
typedef __attribute__((ext_vector_type(8)))  __bf16 v8bf;
typedef __attribute__((ext_vector_type(16))) __bf16 v16bf;
typedef __attribute__((ext_vector_type(8)))  float  v8f;

#define BM 256
#define BN 128
#define BK 32
#define LDS_A 40   // padded row stride (halves): conflict-free b128 reads
#define LDS_B 40

// ---------------------------------------------------------------------------
__global__ __launch_bounds__(256)
void cvt_f32_to_bf16(const float* __restrict__ in, bf16* __restrict__ out, int n) {
  int i = (blockIdx.x * 256 + threadIdx.x) * 8;
  if (i + 8 > n) return;
  float4 a = *(const float4*)(in + i);
  float4 b = *(const float4*)(in + i + 4);
  v8bf o;
  o[0] = (bf16)a.x; o[1] = (bf16)a.y; o[2] = (bf16)a.z; o[3] = (bf16)a.w;
  o[4] = (bf16)b.x; o[5] = (bf16)b.y; o[6] = (bf16)b.z; o[7] = (bf16)b.w;
  *(v8bf*)(out + i) = o;
}

// ---------------------------------------------------------------------------
// helpers: global->reg staging, reg->LDS staging, fragment assembly
// ---------------------------------------------------------------------------
__device__ __forceinline__ void load_gA(const bf16* __restrict__ Ab, int tileM,
                                        int aRow0, int aChunk, int kk, int lda,
                                        v8bf ar[4]) {
#pragma unroll
  for (int i = 0; i < 4; ++i)   // 4 lanes -> 64B contiguous: coalesced
    ar[i] = *(const v8bf*)(Ab + (long long)(tileM + aRow0 + i * 64) * lda + kk + aChunk);
}

__device__ __forceinline__ void load_gB(const bf16* __restrict__ Bb, int tileN,
                                        int bK, int bCol, int kk, int ldb,
                                        v4bf br[4]) {
#pragma unroll
  for (int j = 0; j < 4; ++j)   // lanes -> 8B stride: 256B contiguous per row
    br[j] = *(const v4bf*)(Bb + (long long)(kk + bK + j) * ldb + tileN + bCol);
}

__device__ __forceinline__ void store_sA(bf16* s, int aRow0, int aChunk,
                                         const v8bf ar[4]) {
#pragma unroll
  for (int i = 0; i < 4; ++i)
    *(v8bf*)(&s[(aRow0 + i * 64) * LDS_A + aChunk]) = ar[i];
}

__device__ __forceinline__ void store_sB(bf16* s, int bK, int bCol,
                                         const v4bf br[4]) {
#pragma unroll
  for (int j = 0; j < 4; ++j) {  // 4x4 register transpose -> Bt[n][k]
    v4bf c;
    c[0] = br[0][j]; c[1] = br[1][j]; c[2] = br[2][j]; c[3] = br[3][j];
    *(v4bf*)(&s[(bCol + j) * LDS_B + bK]) = c;
  }
}

__device__ __forceinline__ v16bf frag16(const bf16* lo, const bf16* hi) {
  v8bf l = *(const v8bf*)lo;
  v8bf h = *(const v8bf*)hi;
  return __builtin_shufflevector(l, h, 0,1,2,3,4,5,6,7,8,9,10,11,12,13,14,15);
}

// ---------------------------------------------------------------------------
// Tiled bf16 GEMM, f32 accumulate. out[b] = act(A[b] @ B[b] + bias).
// A: [M x K] row-major, B: [K x N] row-major. Block tile 256x128, BK=32.
// 8 waves (4x2), each wave 64x64 (16 WMMA tiles). Double-buffered LDS.
// Exactly one of outF / outH is non-null.
// ---------------------------------------------------------------------------
__global__ __launch_bounds__(256)
void gemm_bf16_wmma(const bf16* __restrict__ Ag, const bf16* __restrict__ Bg,
                    const float* __restrict__ bias,
                    float* __restrict__ outF, bf16* __restrict__ outH,
                    int N, int K, int lda, int ldb,
                    long long strideA, long long strideB, long long strideO,
                    int relu)
{
  __shared__ bf16 As[2][BM * LDS_A];   // 2 x 256x40 halves = 40960 B
  __shared__ bf16 Bt[2][BN * LDS_B];   // 2 x 128x40 halves = 20480 B (transposed)

  const long long batch = blockIdx.z;
  const bf16* Ab = Ag + batch * strideA;
  const bf16* Bb = Bg + batch * strideB;
  const long long obase = batch * strideO;

  const int tileM = blockIdx.x * BM;
  const int tileN = blockIdx.y * BN;

  const int tid     = threadIdx.x;
  const int lane    = tid & 31;
  const int wave    = tid >> 5;
  const int waveM   = (wave >> 1) * 64;   // 0,64,128,192
  const int waveN   = (wave & 1) * 64;    // 0,64
  const int laneRow = lane & 15;
  const int laneHi  = lane >> 4;          // 0 or 1

  // A staging: 256 rows x 32 halves / 256 thr -> four 16B chunks each
  const int aRow0  = tid >> 2;            // 0..63 (+ i*64)
  const int aChunk = (tid & 3) * 8;       // halves 0,8,16,24
  // B staging: 32 k-rows x 128 cols, 4x4 block transpose
  const int bK   = (tid >> 5) * 4;        // 0..28 (uniform per wave)
  const int bCol = (tid & 31) * 4;        // 0..124

  v8f acc[4][4] = {};
  v8bf ar[4];
  v4bf br[4];

  // prologue: stage first K tile
  load_gA(Ab, tileM, aRow0, aChunk, 0, lda, ar);
  load_gB(Bb, tileN, bK, bCol, 0, ldb, br);
  store_sA(&As[0][0], aRow0, aChunk, ar);
  store_sB(&Bt[0][0], bK, bCol, br);
  __syncthreads();

  int cur = 0;
  for (int kk = 0; kk < K; kk += BK) {
    const bool hasNext = (kk + BK) < K;
    if (hasNext) {            // issue next-tile global loads before compute
      load_gA(Ab, tileM, aRow0, aChunk, kk + BK, lda, ar);
      load_gB(Bb, tileN, bK, bCol, kk + BK, ldb, br);
    }

    // ---- B fragments: col = waveN+ni*16+laneRow, K = laneHi*8 + e (e<8), 16 + laneHi*8 + (e-8) (e>=8) ----
    const bf16* sB = &Bt[cur][0];
    v16bf bfrag[4];
#pragma unroll
    for (int ni = 0; ni < 4; ++ni) {
      const bf16* p = &sB[(waveN + ni * 16 + laneRow) * LDS_B + laneHi * 8];
      bfrag[ni] = frag16(p, p + 16);
    }

    // ---- A fragments + 16 WMMA ----
    const bf16* sA = &As[cur][0];
#pragma unroll
    for (int mi = 0; mi < 4; ++mi) {
      const bf16* p = &sA[(waveM + mi * 16 + laneRow) * LDS_A + laneHi * 8];
      v16bf afrag = frag16(p, p + 16);
#pragma unroll
      for (int ni = 0; ni < 4; ++ni) {
        acc[mi][ni] = __builtin_amdgcn_wmma_f32_16x16x32_bf16(
            false, afrag, false, bfrag[ni],
            (short)0, acc[mi][ni], false, false);
      }
    }

    if (hasNext) {            // fill the other buffer, single barrier
      store_sA(&As[cur ^ 1][0], aRow0, aChunk, ar);
      store_sB(&Bt[cur ^ 1][0], bK, bCol, br);
      __syncthreads();
      cur ^= 1;
    }
  }

  // ---- epilogue: C/D layout VGPR r <-> M = r + 8*laneHi, N = laneRow ----
#pragma unroll
  for (int ni = 0; ni < 4; ++ni) {
    const int col = tileN + waveN + ni * 16 + laneRow;
    const float bv = bias ? bias[col] : 0.0f;
#pragma unroll
    for (int mi = 0; mi < 4; ++mi) {
      const int row0 = tileM + waveM + mi * 16 + laneHi * 8;
#pragma unroll
      for (int r = 0; r < 8; ++r) {
        float v = acc[mi][ni][r] + bv;
        if (relu) v = fmaxf(v, 0.0f);
        const long long idx = obase + (long long)(row0 + r) * N + col;
        if (outF) outF[idx] = v;
        else      outH[idx] = (bf16)v;
      }
    }
  }
}

// ---------------------------------------------------------------------------
// Launch: 3 layers of (S = X@W) then (Y = act(A@S + b)).
// Workspace (~136 MB): adjH (67MB) + x0 (33.5MB) + sH (33.5MB) + weights.
// x0 re-derived from inputs each call, then reused as the layer ping buffer.
// ---------------------------------------------------------------------------
extern "C" void kernel_launch(void* const* d_in, const int* in_sizes, int n_in,
                              void* d_out, int out_size, void* d_ws, size_t ws_size,
                              hipStream_t stream) {
  constexpr int B = 32, N = 1024, F = 512;

  const float* x   = (const float*)d_in[0];   // [B, N, F]
  const float* adj = (const float*)d_in[1];   // [B, N, N]
  const float* W1  = (const float*)d_in[2];
  const float* b1  = (const float*)d_in[3];
  const float* W2  = (const float*)d_in[4];
  const float* b2  = (const float*)d_in[5];
  const float* W3  = (const float*)d_in[6];
  const float* b3  = (const float*)d_in[7];
  float* out = (float*)d_out;

  char* p = (char*)d_ws;
  bf16* adjH = (bf16*)p; p += (size_t)B * N * N * sizeof(bf16);
  bf16* x0   = (bf16*)p; p += (size_t)B * N * F * sizeof(bf16);
  bf16* sH   = (bf16*)p; p += (size_t)B * N * F * sizeof(bf16);
  bf16* w1h  = (bf16*)p; p += (size_t)F * F * sizeof(bf16);
  bf16* w2h  = (bf16*)p; p += (size_t)F * F * sizeof(bf16);
  bf16* w3h  = (bf16*)p; p += (size_t)F * F * sizeof(bf16);
  (void)ws_size; (void)in_sizes; (void)n_in; (void)out_size;

  auto cvt = [&](const float* src, bf16* dst, int n) {
    cvt_f32_to_bf16<<<n / (8 * 256), 256, 0, stream>>>(src, dst, n);
  };
  cvt(adj, adjH, B * N * N);
  cvt(x,   x0,   B * N * F);
  cvt(W1,  w1h,  F * F);
  cvt(W2,  w2h,  F * F);
  cvt(W3,  w3h,  F * F);

  dim3 blk(256);
  dim3 gW((B * N) / BM, F / BN, 1);   // X@W : M = 32768
  dim3 gA(N / BM, F / BN, B);         // A@S : per-graph M = 1024, 32 batches

  const long long sAdj = (long long)N * N;
  const long long sX   = (long long)N * F;

  // Layer 1
  gemm_bf16_wmma<<<gW, blk, 0, stream>>>(x0,   w1h, nullptr, nullptr, sH,
      F, F, F, F, 0, 0, 0, 0);
  gemm_bf16_wmma<<<gA, blk, 0, stream>>>(adjH, sH,  b1, nullptr, x0,
      F, N, N, F, sAdj, sX, sX, 1);
  // Layer 2
  gemm_bf16_wmma<<<gW, blk, 0, stream>>>(x0,   w2h, nullptr, nullptr, sH,
      F, F, F, F, 0, 0, 0, 0);
  gemm_bf16_wmma<<<gA, blk, 0, stream>>>(adjH, sH,  b2, nullptr, x0,
      F, N, N, F, sAdj, sX, sX, 1);
  // Layer 3 (no relu, f32 output)
  gemm_bf16_wmma<<<gW, blk, 0, stream>>>(x0,   w3h, nullptr, nullptr, sH,
      F, F, F, F, 0, 0, 0, 0);
  gemm_bf16_wmma<<<gA, blk, 0, stream>>>(adjH, sH,  b3, out, nullptr,
      F, N, N, F, sAdj, sX, sX, 0);
}
